// CuteInferLinear_82523501626093
// MI455X (gfx1250) — 8-bit matrix operand order repaired and checked by reading, compiled, not run on hardware
//
#include <hip/hip_runtime.h>
#include <hip/hip_bf16.h>


typedef __attribute__((ext_vector_type(16))) int   v16i;
typedef __attribute__((ext_vector_type(8)))  float v8f;
typedef __attribute__((ext_vector_type(4)))  int   v4i;
typedef __attribute__((ext_vector_type(2)))  int   v2i;
typedef int g4i __attribute__((vector_size(16)));

#define CS   128
#define TM   64
#define TN   128
#define NT   128
#define QMAX 448.0f

#if defined(__has_builtin)
#if __has_builtin(__builtin_amdgcn_global_load_async_to_lds_b128)
#define HAVE_ASYNC_LDS 1
#endif
#if __has_builtin(__builtin_amdgcn_cvt_pk_fp8_f32)
#define HAVE_CVT_FP8 1
#endif
#endif

#if defined(HAVE_ASYNC_LDS)
__device__ __forceinline__ void async_cp16(void* lds, const void* gsrc) {
    __builtin_amdgcn_global_load_async_to_lds_b128(
        (__attribute__((address_space(1))) g4i*)(g4i*)gsrc,
        (__attribute__((address_space(3))) g4i*)(g4i*)lds,
        0, 0);
}
__device__ __forceinline__ void wait_async0() {
#if __has_builtin(__builtin_amdgcn_s_wait_asynccnt)
    __builtin_amdgcn_s_wait_asynccnt(0);
#else
    asm volatile("s_wait_asynccnt 0x0" ::: "memory");
#endif
}
#endif

__device__ __forceinline__ unsigned int f32_to_e4m3(float x) {
    unsigned int u = __float_as_uint(x);
    unsigned int s = (u >> 31) << 7;
    float a = fabsf(x);
    if (a >= QMAX) return s | 0x7Eu;
    int e = (int)((u >> 23) & 0xFFu) - 127;
    if (e < -6) {
        float t = a * 512.0f;
        int q = (int)rintf(t);
        return s | (unsigned int)q;
    }
    unsigned int mant = u & 0x7FFFFFu;
    unsigned int keep = mant >> 20;
    unsigned int rest = mant & 0xFFFFFu;
    keep += (rest > 0x80000u) || ((rest == 0x80000u) && (keep & 1u));
    if (keep == 8u) { keep = 0u; ++e; }
    unsigned int code = ((unsigned int)(e + 7) << 3) | keep;
    if (code > 0x7Eu) code = 0x7Eu;
    return s | code;
}

__global__ __launch_bounds__(256) void quant_bf16_to_fp8(
    const __hip_bfloat16* __restrict__ in, unsigned char* __restrict__ q,
    float* __restrict__ scales, int R, int K) {
    const int G     = K / CS;
    const int chunk = blockIdx.x * 8 + (threadIdx.x >> 5);
    const int lane  = threadIdx.x & 31;
    if (chunk >= R * G) return;
    const int row = chunk / G;
    const int g   = chunk % G;

    const size_t base = (size_t)row * K + (size_t)g * CS + (size_t)lane * 4;
    const uint2  raw  = *(const uint2*)((const unsigned short*)in + base);
    float f0 = __uint_as_float((raw.x & 0xFFFFu) << 16);
    float f1 = __uint_as_float(raw.x & 0xFFFF0000u);
    float f2 = __uint_as_float((raw.y & 0xFFFFu) << 16);
    float f3 = __uint_as_float(raw.y & 0xFFFF0000u);

    float amax = fmaxf(fmaxf(fabsf(f0), fabsf(f1)), fmaxf(fabsf(f2), fabsf(f3)));
#pragma unroll
    for (int m = 16; m >= 1; m >>= 1)
        amax = fmaxf(amax, __shfl_xor(amax, m, 32));
    amax = fmaxf(amax, 1e-4f);
    const float inv = QMAX / amax;

#if defined(HAVE_CVT_FP8)
    int pk = 0;
    pk = __builtin_amdgcn_cvt_pk_fp8_f32(f0 * inv, f1 * inv, pk, false);
    pk = __builtin_amdgcn_cvt_pk_fp8_f32(f2 * inv, f3 * inv, pk, true);
    const unsigned int pack = (unsigned int)pk;
#else
    const unsigned int pack =  f32_to_e4m3(f0 * inv)
                            | (f32_to_e4m3(f1 * inv) << 8)
                            | (f32_to_e4m3(f2 * inv) << 16)
                            | (f32_to_e4m3(f3 * inv) << 24);
#endif
    *(unsigned int*)(q + base) = pack;
    if (lane == 0) scales[(size_t)row * G + g] = amax / QMAX;
}

__global__ __launch_bounds__(NT) void gemm_fp8_blockscale(
    const unsigned char* __restrict__ xq, const float* __restrict__ xs,
    const unsigned char* __restrict__ wq, const float* __restrict__ ws,
    const __hip_bfloat16* __restrict__ bias, __hip_bfloat16* __restrict__ out,
    int M, int N, int K) {
    const int G = K / CS;

    __shared__ __align__(16) unsigned char As[2][TM][CS];
    __shared__ __align__(16) unsigned char Bs[2][TN][CS];
    __shared__ float sAs[2][TM];
    __shared__ float sBs[2][TN];

    const int tid  = threadIdx.x;
    const int lane = tid & 31;
    const int wave = tid >> 5;
    const int half = lane >> 4;
    const int l16  = lane & 15;
    const int bm   = blockIdx.y * TM;
    const int bn   = blockIdx.x * TN;
    const int wm   = (wave >> 1) * 32;
    const int wn   = (wave & 1) * 64;

    v8f acc[2][4];
#pragma unroll
    for (int ti = 0; ti < 2; ++ti)
#pragma unroll
        for (int tj = 0; tj < 4; ++tj)
#pragma unroll
            for (int i = 0; i < 8; ++i) acc[ti][tj][i] = 0.0f;

    {
#pragma unroll
        for (int r = 0; r < 4; ++r) {
            int seg = tid + r * NT, row = seg >> 3, col = (seg & 7) << 4;
            const void* src = xq + (size_t)(bm + row) * K + col;
#if defined(HAVE_ASYNC_LDS)
            async_cp16(&As[0][row][col], src);
#else
            *(v4i*)&As[0][row][col] = *(const v4i*)src;
#endif
        }
#pragma unroll
        for (int r = 0; r < 8; ++r) {
            int seg = tid + r * NT, row = seg >> 3, col = (seg & 7) << 4;
            const void* src = wq + (size_t)(bn + row) * K + col;
#if defined(HAVE_ASYNC_LDS)
            async_cp16(&Bs[0][row][col], src);
#else
            *(v4i*)&Bs[0][row][col] = *(const v4i*)src;
#endif
        }
        if (tid < TM) sAs[0][tid] = xs[(size_t)(bm + tid) * G];
        sBs[0][tid] = ws[(size_t)(bn + tid) * G];
#if defined(HAVE_ASYNC_LDS)
        wait_async0();
#endif
    }
    __syncthreads();

    int buf = 0;
    for (int g = 0; g < G; ++g) {
        const bool has_next = (g + 1) < G;
#if defined(HAVE_ASYNC_LDS)
        if (has_next) {
            const size_t koff = (size_t)(g + 1) * CS;
            const int nbuf = buf ^ 1;
#pragma unroll
            for (int r = 0; r < 4; ++r) {
                int seg = tid + r * NT, row = seg >> 3, col = (seg & 7) << 4;
                async_cp16(&As[nbuf][row][col],
                           xq + (size_t)(bm + row) * K + koff + col);
            }
#pragma unroll
            for (int r = 0; r < 8; ++r) {
                int seg = tid + r * NT, row = seg >> 3, col = (seg & 7) << 4;
                async_cp16(&Bs[nbuf][row][col],
                           wq + (size_t)(bn + row) * K + koff + col);
            }
            if (tid < TM) sAs[nbuf][tid] = xs[(size_t)(bm + tid) * G + g + 1];
            sBs[nbuf][tid] = ws[(size_t)(bn + tid) * G + g + 1];
        }
#else
        v4i  na[4], nb[8];
        float nsa = 0.0f, nsb = 0.0f;
        if (has_next) {
            const size_t koff = (size_t)(g + 1) * CS;
#pragma unroll
            for (int r = 0; r < 4; ++r) {
                int seg = tid + r * NT, row = seg >> 3, col = (seg & 7) << 4;
                na[r] = *(const v4i*)(xq + (size_t)(bm + row) * K + koff + col);
            }
#pragma unroll
            for (int r = 0; r < 8; ++r) {
                int seg = tid + r * NT, row = seg >> 3, col = (seg & 7) << 4;
                nb[r] = *(const v4i*)(wq + (size_t)(bn + row) * K + koff + col);
            }
            if (tid < TM) nsa = xs[(size_t)(bm + tid) * G + g + 1];
            nsb = ws[(size_t)(bn + tid) * G + g + 1];
        }
#endif
        if (g + 2 < G) {
            const size_t koff2 = (size_t)(g + 2) * CS;
            __builtin_prefetch(xq + (size_t)(bm + (tid >> 1)) * K + koff2, 0, 0);
            __builtin_prefetch(wq + (size_t)(bn + tid) * K + koff2, 0, 0);
        }

        v16i afrag[2];
#pragma unroll
        for (int ti = 0; ti < 2; ++ti) {
            const unsigned char* ap = &As[buf][wm + ti * 16 + l16][0];
            v2i* av = (v2i*)&afrag[ti];
#pragma unroll
            for (int j = 0; j < 8; ++j)
                av[j] = *(const v2i*)(ap + (j >> 1) * 32 + half * 16 + (j & 1) * 8);
        }
        float sa[2][8];
#pragma unroll
        for (int ti = 0; ti < 2; ++ti)
#pragma unroll
            for (int i = 0; i < 8; ++i)
                sa[ti][i] = sAs[buf][wm + ti * 16 + half * 8 + i];

        const v8f zero = {0.f, 0.f, 0.f, 0.f, 0.f, 0.f, 0.f, 0.f};
#pragma unroll
        for (int tj = 0; tj < 4; ++tj) {
            v16i bfrag;
            const unsigned char* bp = &Bs[buf][wn + tj * 16 + l16][0];
            v4i* bv = (v4i*)&bfrag;
#pragma unroll
            for (int qd = 0; qd < 4; ++qd)
                bv[qd] = *(const v4i*)(bp + qd * 32 + half * 16);
            const float sb = sBs[buf][wn + tj * 16 + l16];
#pragma unroll
            for (int ti = 0; ti < 2; ++ti) {
                v8f p = __builtin_amdgcn_wmma_f32_16x16x128_fp8_fp8(
                    afrag[ti], bfrag, (short)0, zero, false, false);
#pragma unroll
                for (int i = 0; i < 8; ++i)
                    acc[ti][tj][i] += p[i] * (sa[ti][i] * sb);
            }
        }

#if defined(HAVE_ASYNC_LDS)
        if (has_next) wait_async0();
#else
        if (has_next) {
            const int nb_buf = buf ^ 1;
#pragma unroll
            for (int r = 0; r < 4; ++r) {
                int seg = tid + r * NT, row = seg >> 3, col = (seg & 7) << 4;
                *(v4i*)&As[nb_buf][row][col] = na[r];
            }
#pragma unroll
            for (int r = 0; r < 8; ++r) {
                int seg = tid + r * NT, row = seg >> 3, col = (seg & 7) << 4;
                *(v4i*)&Bs[nb_buf][row][col] = nb[r];
            }
            if (tid < TM) sAs[nb_buf][tid] = nsa;
            sBs[nb_buf][tid] = nsb;
        }
#endif
        __syncthreads();
        buf ^= 1;
    }

    float bv[4];
#pragma unroll
    for (int tj = 0; tj < 4; ++tj)
        bv[tj] = __bfloat162float(bias[bn + wn + tj * 16 + l16]);
#pragma unroll
    for (int ti = 0; ti < 2; ++ti)
#pragma unroll
        for (int tj = 0; tj < 4; ++tj)
#pragma unroll
            for (int i = 0; i < 8; ++i) {
                const int row = bm + wm + ti * 16 + half * 8 + i;
                const int col = bn + wn + tj * 16 + l16;
                out[(size_t)row * N + col] =
                    __float2bfloat16(acc[ti][tj][i] + bv[tj]);
            }
}

extern "C" void kernel_launch(void* const* d_in, const int* in_sizes, int n_in,
                              void* d_out, int out_size, void* d_ws, size_t ws_size,
                              hipStream_t stream) {
    const __hip_bfloat16* x    = (const __hip_bfloat16*)d_in[0];
    const __hip_bfloat16* W    = (const __hip_bfloat16*)d_in[1];
    const __hip_bfloat16* bias = (const __hip_bfloat16*)d_in[2];

    const int N = in_sizes[2];
    const int K = in_sizes[1] / N;
    const int M = in_sizes[0] / K;
    const int G = K / CS;

    unsigned char* xq = (unsigned char*)d_ws;
    unsigned char* wq = xq + (size_t)M * K;
    float*         xs = (float*)(wq + (size_t)N * K);
    float*         ws = xs + (size_t)M * G;

    {
        int xchunks = M * G, wchunks = N * G;
        quant_bf16_to_fp8<<<(xchunks + 7) / 8, 256, 0, stream>>>(x, xq, xs, M, K);
        quant_bf16_to_fp8<<<(wchunks + 7) / 8, 256, 0, stream>>>(W, wq, ws, N, K);
    }
    dim3 grid(N / TN, M / TM);
    gemm_fp8_blockscale<<<grid, NT, 0, stream>>>(xq, xs, wq, ws, bias,
                                                 (__hip_bfloat16*)d_out, M, N, K);
}
